// LinTrans_78168404787748
// MI455X (gfx1250) — matrix operand order repaired and checked by reading, not compiled, not run on hardware
//
#include <hip/hip_runtime.h>
#include <math.h>

// ---------------------------------------------------------------------------
// Types / helpers
// ---------------------------------------------------------------------------
typedef float v8f  __attribute__((ext_vector_type(8)));
typedef __bf16 v16bf __attribute__((ext_vector_type(16)));

union ABFrag { v16bf v; unsigned int u[8]; };

__device__ __forceinline__ unsigned short f2bf(float f) {
  unsigned int u = __float_as_uint(f);
  unsigned int r = u + 0x7FFFu + ((u >> 16) & 1u);   // round-to-nearest-even
  return (unsigned short)(r >> 16);
}
__device__ __forceinline__ float bf2f(unsigned short h) {
  return __uint_as_float(((unsigned int)h) << 16);
}
// A-fragment K-pair base for 16-bit 16x32 A layout (lane = half*16 + m)
__device__ __forceinline__ int akb(int v, int half) {
  return (v < 4 ? (v << 1) : 16 + ((v - 4) << 1)) + (half << 3);
}

// CDNA5 async global->LDS copy (ASYNCcnt-tracked DMA; 16B per lane).
// lds_off is a byte offset from the wave's LDS base (dynamic LDS starts at
// groupstaticsize()); gptr is a per-lane global address.
__device__ __forceinline__ void async_g2l_b128(unsigned lds_off, const void* gptr) {
  asm volatile("global_load_async_to_lds_b128 %0, %1, off"
               :: "v"(lds_off), "v"(gptr) : "memory");
}
__device__ __forceinline__ void wait_async0() {
  asm volatile("s_wait_asynccnt 0" ::: "memory");
}

#define BST 34   // LDS row stride (elements) for GEMM transposed B panels
#define BKST 40  // local-attn k-panel row stride (80B, 16B-aligned for async b128)

// ---------------------------------------------------------------------------
// Generic bf16 WMMA GEMM:  C[M,N] = A[M,K](bf16,rm) * B[K,N](bf16,rm) + epilogue
// Block: 256 threads = 8 waves (4 in M x 2 in N). Each wave: 16 rows x NT tiles.
// grid.z = batch; strides in elements (0 => shared across batch).
// ---------------------------------------------------------------------------
template <int NT>
__global__ __launch_bounds__(256) void gemm_wmma(
    const unsigned short* __restrict__ A, const unsigned short* __restrict__ B,
    float* Cf, unsigned short* Cb,
    int M, int N, int K,
    long long sA, long long sB, long long sC,
    const float* __restrict__ bias_row, const float* __restrict__ bias_col,
    const float* __restrict__ add_mat, const float* residual, int act)
{
  extern __shared__ unsigned short gemm_smem[];   // [N][BST] bf16, transposed panel
  const int tid  = threadIdx.x;
  const int lane = tid & 31, wid = tid >> 5;
  const int half = lane >> 4, l16 = lane & 15;
  const int wm = wid & 3, wn = wid >> 2;
  const int z = blockIdx.z;
  const unsigned short* Ab = A + (size_t)z * (size_t)sA;
  const unsigned short* Bb = B + (size_t)z * (size_t)sB;
  const size_t coff = (size_t)z * (size_t)sC;

  const int mbase = blockIdx.x * 64 + wm * 16;
  const int m = mbase + l16;

  const v8f vzero = {0.f,0.f,0.f,0.f,0.f,0.f,0.f,0.f};
  v8f acc[NT];
#pragma unroll
  for (int t = 0; t < NT; ++t) acc[t] = vzero;

  if (m < M) __builtin_prefetch(Ab + (size_t)m * K, 0, 3);

  for (int k0 = 0; k0 < K; k0 += 32) {
    __syncthreads();
    for (int idx = tid; idx < (N << 5); idx += 256) {
      int n = idx % N, kk = idx / N;
      int kg = k0 + kk;
      unsigned short vsh = 0;
      if (kg < K) vsh = Bb[(size_t)kg * N + n];
      gemm_smem[n * BST + kk] = vsh;
    }
    __syncthreads();

    ABFrag af;
    if (m < M) {
      const unsigned short* ar = Ab + (size_t)m * K + k0;
#pragma unroll
      for (int v = 0; v < 8; ++v)
        af.u[v] = *(const unsigned int*)(ar + akb(v, half));
    } else {
#pragma unroll
      for (int v = 0; v < 8; ++v) af.u[v] = 0u;
    }

#pragma unroll
    for (int t = 0; t < NT; ++t) {
      int n0 = (wn + (t << 1)) << 4;
      ABFrag bfr;
      const unsigned short* br = gemm_smem + (n0 + l16) * BST;
#pragma unroll
      for (int v = 0; v < 8; ++v)
        bfr.u[v] = *(const unsigned int*)(br + akb(v, half));
      acc[t] = __builtin_amdgcn_wmma_f32_16x16x32_bf16(
          false, af.v, false, bfr.v, (short)0, acc[t], false, false);
    }
  }

#pragma unroll
  for (int t = 0; t < NT; ++t) {
    int n0 = (wn + (t << 1)) << 4;
    int n = n0 + l16;
    if (n >= N) continue;
    float bcv = bias_col ? bias_col[n] : 0.0f;
#pragma unroll
    for (int v = 0; v < 8; ++v) {
      int mm = mbase + v + (half << 3);
      if (mm >= M) continue;
      float val = acc[t][v] + bcv;
      if (bias_row)  val += bias_row[mm];
      if (add_mat)   val += add_mat[(size_t)mm * N + n];
      if (residual)  val += residual[coff + (size_t)mm * N + n];
      if (act == 1)  val = 0.5f * val * (1.0f + erff(val * 0.70710678118654752f));
      size_t o = coff + (size_t)mm * N + n;
      if (Cf) Cf[o] = val;
      if (Cb) Cb[o] = f2bf(val);
    }
  }
}

// ---------------------------------------------------------------------------
// Fused embed: Wc = expand_w @ emb_w (5x64), bc = expand_b @ emb_w + emb_b
// ---------------------------------------------------------------------------
__global__ void fuse_embed_kernel(const float* __restrict__ ew, const float* __restrict__ eb,
                                  const float* __restrict__ mw, const float* __restrict__ mb,
                                  float* __restrict__ Wc, float* __restrict__ bc)
{
  int d = threadIdx.x;
  if (d >= 64) return;
  for (int j = 0; j < 5; ++j) {
    float s = 0.0f;
    for (int c = 0; c < 64; ++c) s += ew[j * 64 + c] * mw[c * 64 + d];
    Wc[j * 64 + d] = s;
  }
  float s = mb[d];
  for (int c = 0; c < 64; ++c) s += eb[c] * mw[c * 64 + d];
  bc[d] = s;
}

__global__ void embed_kernel(const float* __restrict__ x, const float* __restrict__ Wc,
                             const float* __restrict__ bc, unsigned short* __restrict__ h0,
                             int total)
{
  int idx = blockIdx.x * 256 + threadIdx.x;
  if (idx >= total) return;
  int d = idx & 63;
  int row = idx >> 6;
  const float* xr = x + row * 5;
  float s = bc[d];
#pragma unroll
  for (int j = 0; j < 5; ++j) s += xr[j] * Wc[j * 64 + d];
  h0[idx] = f2bf(s);
}

// conv_w (O=1024, I=1024, k=3) -> A[o, k*1024+i] bf16 (K = 3072)
__global__ void repack_convA_kernel(const float* __restrict__ w, unsigned short* __restrict__ A)
{
  int idx = blockIdx.x * 256 + threadIdx.x;
  if (idx >= 1024 * 3072) return;
  int o = idx / 3072;
  int r = idx - o * 3072;
  int k = r >> 10, i = r & 1023;
  A[idx] = f2bf(w[(o * 1024 + i) * 3 + k]);
}

// B_b[(k*1024+i), d] = h0[b, i, d + k - 1] (zero pad along d)
__global__ void build_convB_kernel(const unsigned short* __restrict__ h0,
                                   unsigned short* __restrict__ Bm)
{
  long long idx = (long long)blockIdx.x * 256 + threadIdx.x;
  if (idx >= (long long)64 * 3072 * 64) return;
  int d = (int)(idx & 63);
  long long r = idx >> 6;               // b*3072 + k*1024 + i
  int ki = (int)(r % 3072);
  int b  = (int)(r / 3072);
  int k = ki >> 10, i = ki & 1023;
  int ds = d + k - 1;
  unsigned short v = 0;
  if (ds >= 0 && ds < 64) v = h0[((size_t)b * 1024 + i) * 64 + ds];
  Bm[idx] = v;
}

// pack wq|wk|wv (each DEPTH x 64 x 64) -> [layer][c][192] bf16
__global__ void pack_qkv_kernel(const float* __restrict__ wq, const float* __restrict__ wk,
                                const float* __restrict__ wv, unsigned short* __restrict__ dst)
{
  int idx = blockIdx.x * 256 + threadIdx.x;
  if (idx >= 2 * 64 * 192) return;
  int i = idx / (64 * 192);
  int r = idx - i * (64 * 192);
  int c = r / 192, j = r - c * 192;
  int sel = j >> 6, d = j & 63;
  const float* src = (sel == 0) ? wq : ((sel == 1) ? wk : wv);
  dst[idx] = f2bf(src[(i * 64 + c) * 64 + d]);
}

__global__ void f32_to_bf16_kernel(const float* __restrict__ src,
                                   unsigned short* __restrict__ dst, int n)
{
  int idx = blockIdx.x * 256 + threadIdx.x;
  if (idx < n) dst[idx] = f2bf(src[idx]);
}

// ---------------------------------------------------------------------------
// LayerNorm over D=64: one wave (32 lanes) per row, shuffle reductions, bf16 out
// ---------------------------------------------------------------------------
__global__ __launch_bounds__(256) void ln_kernel(const float* __restrict__ h,
    const float* __restrict__ sc, const float* __restrict__ bi,
    unsigned short* __restrict__ outp, int nrows)
{
  int wid = threadIdx.x >> 5, lane = threadIdx.x & 31;
  int row = blockIdx.x * 8 + wid;
  if (row >= nrows) return;
  const float* r = h + (size_t)row * 64;
  float a0 = r[lane], a1 = r[lane + 32];
  float s = a0 + a1;
#pragma unroll
  for (int o = 16; o > 0; o >>= 1) s += __shfl_xor(s, o, 32);
  float mu = s * (1.0f / 64.0f);
  float d0 = a0 - mu, d1 = a1 - mu;
  float vs = d0 * d0 + d1 * d1;
#pragma unroll
  for (int o = 16; o > 0; o >>= 1) vs += __shfl_xor(vs, o, 32);
  float rstd = rsqrtf(vs * (1.0f / 64.0f) + 1e-5f);
  outp[(size_t)row * 64 + lane]      = f2bf(d0 * rstd * sc[lane] + bi[lane]);
  outp[(size_t)row * 64 + lane + 32] = f2bf(d1 * rstd * sc[lane + 32] + bi[lane + 32]);
}

// ---------------------------------------------------------------------------
// Local windowed attention: one WG per (b, head<4, window). 8 waves.
// q/k rows DMA'd global->LDS with global_load_async_to_lds_b128 (ASYNCcnt),
// dots (128x384) via WMMA (e padded 8->32), fp32 softmax in LDS, P@V via WMMA.
// ---------------------------------------------------------------------------
#define LA_DOTS_OFF 0                            // 128*384*4 = 196608
#define LA_BQ_OFF   196608                       // 128 rows * 64B (32 bf16)
#define LA_BK_OFF   (LA_BQ_OFF + 128 * 64)       // 204800; 384 rows * 80B (BKST)
#define LA_BV_OFF   (LA_BK_OFF + 384 * BKST * 2) // 235520; 16 rows * 784B
#define LA_SMEM_BYTES (LA_BV_OFF + 16 * 392 * 2) // 248064 (<= 320KB WGP LDS)

__global__ __launch_bounds__(256) void local_attn_kernel(const unsigned short* __restrict__ qkvb,
                                                         unsigned short* __restrict__ attnb)
{
  extern __shared__ char la_smem[];
  float* dots        = (float*)(la_smem + LA_DOTS_OFF);        // [128][384]
  unsigned short* bq = (unsigned short*)(la_smem + LA_BQ_OFF); // [128][32]
  unsigned short* bk = (unsigned short*)(la_smem + LA_BK_OFF); // [384][BKST]
  unsigned short* bv = (unsigned short*)(la_smem + LA_BV_OFF); // [16][392]

  int bx = blockIdx.x;
  int b  = bx >> 5;
  int hh = (bx >> 3) & 3;
  int wi = bx & 7;
  int tid = threadIdx.x;
  int lane = tid & 31, wid = tid >> 5;
  int half = lane >> 4, l16 = lane & 15;
  const size_t rowbase = (size_t)b * 1024;
  const unsigned short* qb = qkvb + rowbase * 192;
  const unsigned ldsbase = __builtin_amdgcn_groupstaticsize(); // dynamic LDS start

  // --- async DMA: q rows (16B each: e=0..7) straight into LDS -------------
  if (wid < 4) {
    int row = wid * 32 + lane;
    async_g2l_b128(ldsbase + LA_BQ_OFF + row * 64,
                   qb + (size_t)(wi * 128 + row) * 192 + hh * 8);
  }
  // --- async DMA: k rows; window validity is 32-row aligned => wave-uniform
  {
    int j = tid;                       // rows 0..255
    int ts = (wi - 1) * 128 + j;
    if (ts >= 0 && ts < 1024)
      async_g2l_b128(ldsbase + LA_BK_OFF + j * (BKST * 2),
                     qb + (size_t)ts * 192 + 64 + hh * 8);
    int j2 = 256 + tid;                // rows 256..383 (waves 0..3 full)
    if (j2 < 384) {
      int ts2 = (wi - 1) * 128 + j2;
      if (ts2 >= 0 && ts2 < 1024)
        async_g2l_b128(ldsbase + LA_BK_OFF + j2 * (BKST * 2),
                       qb + (size_t)ts2 * 192 + 64 + hh * 8);
    }
  }
  // zero-fill q pad (e 8..31): 128 rows * 12 dwords
  for (int idx = tid; idx < 128 * 12; idx += 256) {
    int i = idx / 12, c = idx - i * 12;
    *(unsigned int*)(la_smem + LA_BQ_OFF + i * 64 + 16 + c * 4) = 0u;
  }
  // zero-fill k pad (e 8..31): 384 rows * 12 dwords
  for (int idx = tid; idx < 384 * 12; idx += 256) {
    int j = idx / 12, c = idx - j * 12;
    *(unsigned int*)(la_smem + LA_BK_OFF + j * (BKST * 2) + 16 + c * 4) = 0u;
  }
  // zero-fill k data for out-of-sequence rows (not DMA'd)
  for (int idx = tid; idx < 384 * 4; idx += 256) {
    int j = idx / 4, c = idx - j * 4;
    int ts = (wi - 1) * 128 + j;
    if (ts < 0 || ts >= 1024)
      *(unsigned int*)(la_smem + LA_BK_OFF + j * (BKST * 2) + c * 4) = 0u;
  }
  // v^T staging (transposed gather: plain LDS stores), rows e padded 8->16
  for (int idx = tid; idx < 16 * 384; idx += 256) {
    int e = idx / 384, j = idx - e * 384;
    int ts = (wi - 1) * 128 + j;
    unsigned short v = 0;
    if (e < 8 && ts >= 0 && ts < 1024)
      v = qb[(size_t)ts * 192 + 128 + hh * 8 + e];
    bv[e * 392 + j] = v;
  }
  wait_async0();      // drain ASYNCcnt before publishing the panel
  __syncthreads();

  // dots = (q @ k^T) * DH^-0.5, with pad-window mask
  {
    int mt = wid;                                        // 8 waves -> 8 M tiles
    ABFrag af;
    const unsigned short* ar = bq + (mt * 16 + l16) * 32;
#pragma unroll
    for (int v = 0; v < 8; ++v) af.u[v] = *(const unsigned int*)(ar + akb(v, half));
    for (int nt = 0; nt < 24; ++nt) {
      ABFrag bfr;
      const unsigned short* br = bk + (nt * 16 + l16) * BKST;
#pragma unroll
      for (int v = 0; v < 8; ++v)
        bfr.u[v] = *(const unsigned int*)(br + akb(v, half));
      v8f c = {0.f,0.f,0.f,0.f,0.f,0.f,0.f,0.f};
      c = __builtin_amdgcn_wmma_f32_16x16x32_bf16(false, af.v, false, bfr.v,
                                                  (short)0, c, false, false);
      int j = nt * 16 + l16;
      bool maskcol = (wi == 0 && j < 128) || (wi == 7 && j >= 256);
#pragma unroll
      for (int v = 0; v < 8; ++v) {
        int mr = mt * 16 + v + (half << 3);
        float val = c[v] * 0.35355339059327373f;
        if (maskcol) val = -3.0e38f;
        dots[mr * 384 + j] = val;
      }
    }
  }
  __syncthreads();

  // fp32 softmax along j per row
  if (tid < 128) {
    float* dr = dots + tid * 384;
    float mx = -3.4e38f;
    for (int j = 0; j < 384; ++j) mx = fmaxf(mx, dr[j]);
    float sum = 0.0f;
    for (int j = 0; j < 384; ++j) { float e = __expf(dr[j] - mx); dr[j] = e; sum += e; }
    float inv = 1.0f / sum;
    for (int j = 0; j < 384; ++j) dr[j] *= inv;
  }
  __syncthreads();

  // out = P @ V  (K loop over j, N = e padded to 16)
  {
    int mt = wid;
    v8f c = {0.f,0.f,0.f,0.f,0.f,0.f,0.f,0.f};
    for (int j0 = 0; j0 < 384; j0 += 32) {
      ABFrag af;
      const float* dr = dots + (mt * 16 + l16) * 384 + j0;
#pragma unroll
      for (int v = 0; v < 8; ++v) {
        int kk = akb(v, half);
        unsigned int lo = f2bf(dr[kk]);
        unsigned int hi = f2bf(dr[kk + 1]);
        af.u[v] = lo | (hi << 16);
      }
      ABFrag bfr;
      const unsigned short* br = bv + l16 * 392 + j0;
#pragma unroll
      for (int v = 0; v < 8; ++v)
        bfr.u[v] = *(const unsigned int*)(br + akb(v, half));
      c = __builtin_amdgcn_wmma_f32_16x16x32_bf16(false, af.v, false, bfr.v,
                                                  (short)0, c, false, false);
    }
    int e = l16;
    if (e < 8) {
#pragma unroll
      for (int v = 0; v < 8; ++v) {
        int mr = mt * 16 + v + (half << 3);
        int t = wi * 128 + mr;
        attnb[(rowbase + t) * 64 + hh * 8 + e] = f2bf(c[v]);
      }
    }
  }
}

// ---------------------------------------------------------------------------
// Global linear-attention heads (4..7): softmax(q,e)*DH^-0.5, softmax(k,n),
// ctx = k^T v (8x8), out = q ctx.  One WG per (b, head). bf16 inputs.
// ---------------------------------------------------------------------------
__global__ __launch_bounds__(256) void global_attn_kernel(const unsigned short* __restrict__ qkvb,
                                                          unsigned short* __restrict__ attnb)
{
  __shared__ float red[256 * 8];
  __shared__ float colmax[8], colsum[8], ctx[64];
  int b  = blockIdx.x >> 2;
  int hg = 4 + (blockIdx.x & 3);
  int tid = threadIdx.x;
  size_t rowbase = (size_t)b * 1024;
  int qcol = hg * 8, kcol = 64 + hg * 8, vcol = 128 + hg * 8;

  float lmax[8];
#pragma unroll
  for (int d = 0; d < 8; ++d) lmax[d] = -3.4e38f;
  for (int n = tid; n < 1024; n += 256) {
    const unsigned short* r = qkvb + (rowbase + n) * 192 + kcol;
#pragma unroll
    for (int d = 0; d < 8; ++d) lmax[d] = fmaxf(lmax[d], bf2f(r[d]));
  }
#pragma unroll
  for (int d = 0; d < 8; ++d) red[tid * 8 + d] = lmax[d];
  __syncthreads();
  if (tid < 8) {
    float m = -3.4e38f;
    for (int i = 0; i < 256; ++i) m = fmaxf(m, red[i * 8 + tid]);
    colmax[tid] = m;
  }
  __syncthreads();
  float lsum[8];
#pragma unroll
  for (int d = 0; d < 8; ++d) lsum[d] = 0.0f;
  for (int n = tid; n < 1024; n += 256) {
    const unsigned short* r = qkvb + (rowbase + n) * 192 + kcol;
#pragma unroll
    for (int d = 0; d < 8; ++d) lsum[d] += __expf(bf2f(r[d]) - colmax[d]);
  }
#pragma unroll
  for (int d = 0; d < 8; ++d) red[tid * 8 + d] = lsum[d];
  __syncthreads();
  if (tid < 8) {
    float s = 0.0f;
    for (int i = 0; i < 256; ++i) s += red[i * 8 + tid];
    colsum[tid] = s;
  }
  __syncthreads();
  {
    int p = tid & 63, sl = tid >> 6;
    int d = p >> 3, e = p & 7;
    float part = 0.0f;
    for (int n = sl; n < 1024; n += 4) {
      const unsigned short* r = qkvb + (rowbase + n) * 192;
      float gk = __expf(bf2f(r[kcol + d]) - colmax[d]) / colsum[d];
      part += gk * bf2f(r[vcol + e]);
    }
    red[tid] = part;
  }
  __syncthreads();
  if (tid < 64) ctx[tid] = red[tid] + red[tid + 64] + red[tid + 128] + red[tid + 192];
  __syncthreads();
  for (int n = tid; n < 1024; n += 256) {
    const unsigned short* r = qkvb + (rowbase + n) * 192 + qcol;
    float q[8], mx = -3.4e38f;
#pragma unroll
    for (int d = 0; d < 8; ++d) { q[d] = bf2f(r[d]); mx = fmaxf(mx, q[d]); }
    float sum = 0.0f;
#pragma unroll
    for (int d = 0; d < 8; ++d) { q[d] = __expf(q[d] - mx); sum += q[d]; }
    float scq = 0.35355339059327373f / sum;
#pragma unroll
    for (int e = 0; e < 8; ++e) {
      float o = 0.0f;
#pragma unroll
      for (int d = 0; d < 8; ++d) o += q[d] * ctx[d * 8 + e];
      attnb[(rowbase + n) * 64 + hg * 8 + e] = f2bf(o * scq);
    }
  }
}

// Final collapse: out[row, j<5] = h[row,:] @ collapse_w + collapse_b  (fp32)
__global__ void collapse_kernel(const float* __restrict__ h, const float* __restrict__ cw,
                                const float* __restrict__ cb, float* __restrict__ outp,
                                int rows)
{
  int idx = blockIdx.x * 256 + threadIdx.x;
  if (idx >= rows * 5) return;
  int row = idx / 5, j = idx - row * 5;
  const float* hr = h + (size_t)row * 64;
  float s = cb[j];
  for (int d = 0; d < 64; ++d) s += hr[d] * cw[d * 5 + j];
  outp[idx] = s;
}

// ---------------------------------------------------------------------------
// Host orchestration
// ---------------------------------------------------------------------------
extern "C" void kernel_launch(void* const* d_in, const int* in_sizes, int n_in,
                              void* d_out, int out_size, void* d_ws, size_t ws_size,
                              hipStream_t stream)
{
  const float* x        = (const float*)d_in[0];
  const float* pos_emb  = (const float*)d_in[1];
  const float* expand_w = (const float*)d_in[2];
  const float* expand_b = (const float*)d_in[3];
  const float* emb_w    = (const float*)d_in[4];
  const float* emb_b    = (const float*)d_in[5];
  const float* conv_w   = (const float*)d_in[6];
  const float* conv_b   = (const float*)d_in[7];
  const float* ln1_s    = (const float*)d_in[8];
  const float* ln1_b    = (const float*)d_in[9];
  const float* wq       = (const float*)d_in[10];
  const float* wk       = (const float*)d_in[11];
  const float* wv       = (const float*)d_in[12];
  const float* wo       = (const float*)d_in[13];
  const float* wo_b     = (const float*)d_in[14];
  const float* ln2_s    = (const float*)d_in[15];
  const float* ln2_b    = (const float*)d_in[16];
  const float* ff_w1    = (const float*)d_in[17];
  const float* ff_b1    = (const float*)d_in[18];
  const float* ff_w2    = (const float*)d_in[19];
  const float* ff_b2    = (const float*)d_in[20];
  const float* coll_w   = (const float*)d_in[21];
  const float* coll_b   = (const float*)d_in[22];
  float* outp = (float*)d_out;
  (void)in_sizes; (void)n_in; (void)out_size; (void)ws_size;

  char* base = (char*)d_ws;
  size_t off = 0;
  auto walloc = [&](size_t bytes) -> void* {
    void* p = base + off;
    off = (off + bytes + 255) & ~(size_t)255;
    return p;
  };
  const int BT = 64 * 1024;
  float*          Wc    = (float*)walloc(5 * 64 * 4);
  float*          bc    = (float*)walloc(64 * 4);
  unsigned short* h0b   = (unsigned short*)walloc((size_t)BT * 64 * 2);
  unsigned short* convA = (unsigned short*)walloc((size_t)1024 * 3072 * 2);
  unsigned short* convB = (unsigned short*)walloc((size_t)64 * 3072 * 64 * 2);
  float*          hbuf  = (float*)walloc((size_t)BT * 64 * 4);
  unsigned short* ybf   = (unsigned short*)walloc((size_t)BT * 64 * 2);
  unsigned short* qkvb  = (unsigned short*)walloc((size_t)BT * 192 * 2);
  unsigned short* attnb = (unsigned short*)walloc((size_t)BT * 64 * 2);
  unsigned short* gbf   = (unsigned short*)walloc((size_t)BT * 256 * 2);
  unsigned short* wqkvb = (unsigned short*)walloc((size_t)2 * 64 * 192 * 2);
  unsigned short* wob   = (unsigned short*)walloc((size_t)2 * 64 * 64 * 2);
  unsigned short* w1b   = (unsigned short*)walloc((size_t)2 * 64 * 256 * 2);
  unsigned short* w2b   = (unsigned short*)walloc((size_t)2 * 256 * 64 * 2);

  // embed (fused affine 5->64), to bf16
  fuse_embed_kernel<<<1, 64, 0, stream>>>(expand_w, expand_b, emb_w, emb_b, Wc, bc);
  embed_kernel<<<(BT * 64) / 256, 256, 0, stream>>>(x, Wc, bc, h0b, BT * 64);

  // weight repacks to bf16
  repack_convA_kernel<<<(1024 * 3072) / 256, 256, 0, stream>>>(conv_w, convA);
  {
    long long tot = (long long)64 * 3072 * 64;
    build_convB_kernel<<<(unsigned)((tot + 255) / 256), 256, 0, stream>>>(h0b, convB);
  }
  pack_qkv_kernel<<<(2 * 64 * 192 + 255) / 256, 256, 0, stream>>>(wq, wk, wv, wqkvb);
  f32_to_bf16_kernel<<<(2 * 64 * 64 + 255) / 256, 256, 0, stream>>>(wo, wob, 2 * 64 * 64);
  f32_to_bf16_kernel<<<(2 * 64 * 256 + 255) / 256, 256, 0, stream>>>(ff_w1, w1b, 2 * 64 * 256);
  f32_to_bf16_kernel<<<(2 * 256 * 64 + 255) / 256, 256, 0, stream>>>(ff_w2, w2b, 2 * 256 * 64);

  // conv as batched GEMM: per batch (M=1024, K=3072, N=64) + row bias + pos_emb
  {
    dim3 g(1024 / 64, 1, 64);
    gemm_wmma<2><<<g, 256, (size_t)64 * BST * 2, stream>>>(
        convA, convB, hbuf, nullptr, 1024, 64, 3072,
        0LL, (long long)3072 * 64, (long long)1024 * 64,
        conv_b, nullptr, pos_emb, nullptr, 0);
  }

  for (int layer = 0; layer < 2; ++layer) {
    ln_kernel<<<BT / 8, 256, 0, stream>>>(hbuf, ln1_s + layer * 64, ln1_b + layer * 64, ybf, BT);
    // fused QKV GEMM: (BT x 64) @ (64 x 192), bf16 output for attention DMA
    gemm_wmma<6><<<dim3(BT / 64, 1, 1), 256, (size_t)192 * BST * 2, stream>>>(
        ybf, wqkvb + layer * 64 * 192, nullptr, qkvb, BT, 192, 64,
        0LL, 0LL, 0LL, nullptr, nullptr, nullptr, nullptr, 0);

    local_attn_kernel<<<2048, 256, LA_SMEM_BYTES, stream>>>(qkvb, attnb);
    global_attn_kernel<<<256, 256, 0, stream>>>(qkvb, attnb);

    // output proj + residual into hbuf
    gemm_wmma<2><<<dim3(BT / 64, 1, 1), 256, (size_t)64 * BST * 2, stream>>>(
        attnb, wob + layer * 64 * 64, hbuf, nullptr, BT, 64, 64,
        0LL, 0LL, 0LL, nullptr, wo_b + layer * 64, nullptr, hbuf, 0);

    ln_kernel<<<BT / 8, 256, 0, stream>>>(hbuf, ln2_s + layer * 64, ln2_b + layer * 64, ybf, BT);
    // FF1 with exact GELU epilogue, bf16 output
    gemm_wmma<8><<<dim3(BT / 64, 1, 1), 256, (size_t)256 * BST * 2, stream>>>(
        ybf, w1b + layer * 64 * 256, nullptr, gbf, BT, 256, 64,
        0LL, 0LL, 0LL, nullptr, ff_b1 + layer * 256, nullptr, nullptr, 1);
    // FF2 + residual into hbuf
    gemm_wmma<2><<<dim3(BT / 64, 1, 1), 256, (size_t)64 * BST * 2, stream>>>(
        gbf, w2b + layer * 256 * 64, hbuf, nullptr, BT, 64, 256,
        0LL, 0LL, 0LL, nullptr, ff_b2 + layer * 64, nullptr, hbuf, 0);
  }

  collapse_kernel<<<(BT * 5 + 255) / 256, 256, 0, stream>>>(hbuf, coll_w, coll_b, outp, BT);
}
